// ArcticAttention_5549097746649
// MI455X (gfx1250) — matrix operand order repaired and checked by reading, not compiled, not run on hardware
//
#include <hip/hip_runtime.h>
#include <hip/hip_bf16.h>
#include <stdint.h>

// ---------------------------------------------------------------------------
// Problem constants (match the reference)
// ---------------------------------------------------------------------------
static constexpr int Bc   = 2;
static constexpr int Tc   = 2048;
static constexpr int HIDc = 2048;
static constexpr int NHc  = 16;
static constexpr int NKVc = 4;
static constexpr int HDc  = 128;       // HID / NH
static constexpr int WINc = 1024;
static constexpr int REPc = NHc / NKVc;

typedef __attribute__((ext_vector_type(16))) __bf16 v16bf;
typedef __attribute__((ext_vector_type(8)))  float  v8f;

union Frag16 {
    v16bf v;
    unsigned short us[16];
    uint4 q[2];           // two 16-byte halves (8 bf16 each)
};

__device__ __forceinline__ unsigned short f2bf(float f) {
    unsigned int u = __float_as_uint(f);
    unsigned int r = u + 0x7FFFu + ((u >> 16) & 1u);   // round-to-nearest-even
    return (unsigned short)(r >> 16);
}

// ---------------------------------------------------------------------------
// Convert f32 -> bf16 (vectorized by 4)
// ---------------------------------------------------------------------------
__global__ void cvt_f32_to_bf16(const float* __restrict__ x,
                                unsigned short* __restrict__ y, int n4) {
    int i = blockIdx.x * blockDim.x + threadIdx.x;
    if (i >= n4) return;
    float4 f = reinterpret_cast<const float4*>(x)[i];
    ushort4 o;
    o.x = f2bf(f.x); o.y = f2bf(f.y); o.z = f2bf(f.z); o.w = f2bf(f.w);
    reinterpret_cast<ushort4*>(y)[i] = o;
}

// ---------------------------------------------------------------------------
// Register-blocked WMMA GEMM:
//   out[M,N] (f32) = A[M,K] (bf16 row-major) * Wbf[N,K]^T (bf16 row-major)
// One wave computes a 64x64 output block (4x4 grid of 16x16 WMMA tiles).
// Per K-chunk of 32: 4 A-fragment loads + 4 B-fragment loads -> 16 WMMAs.
// ---------------------------------------------------------------------------
__global__ __launch_bounds__(256) void wmma_gemm_nt(
        const unsigned short* __restrict__ A,
        const unsigned short* __restrict__ Wbf,
        float* __restrict__ out,
        int M, int N, int K) {
    const int lane = threadIdx.x & 31;
    const int wave = threadIdx.x >> 5;
    const int tile = blockIdx.x * 8 + wave;
    const int ntn  = N >> 6;                  // 64-wide blocks in N
    const int ntot = (M >> 6) * ntn;
    if (tile >= ntot) return;
    const int tm = tile / ntn;
    const int tn = tile - tm * ntn;
    const int m0 = tm * 64;
    const int n0 = tn * 64;

    const int lo = lane & 15;
    const int hi = lane >> 4;

    // Per-lane row pointers for the 4 M-subtiles / 4 N-subtiles
    const unsigned short* arow[4];
    const unsigned short* brow[4];
#pragma unroll
    for (int i = 0; i < 4; ++i) {
        arow[i] = A   + (size_t)(m0 + i * 16 + lo) * K;
        brow[i] = Wbf + (size_t)(n0 + i * 16 + lo) * K;
    }

    v8f acc[4][4];
#pragma unroll
    for (int i = 0; i < 4; ++i)
#pragma unroll
        for (int j = 0; j < 4; ++j) acc[i][j] = v8f{};

    for (int kb = 0; kb < K; kb += 32) {
        Frag16 a[4], b[4];
        const int ka = kb + hi * 8;      // A layout: runs at ka and ka+16
        const int kw = kb + hi * 8;      // B layout: runs at kw and kw+16
#pragma unroll
        for (int i = 0; i < 4; ++i) {
            a[i].q[0] = *reinterpret_cast<const uint4*>(arow[i] + ka);
            a[i].q[1] = *reinterpret_cast<const uint4*>(arow[i] + ka + 16);
        }
#pragma unroll
        for (int j = 0; j < 4; ++j) {
            b[j].q[0] = *reinterpret_cast<const uint4*>(brow[j] + kw);
            b[j].q[1] = *reinterpret_cast<const uint4*>(brow[j] + kw + 16);
        }
#pragma unroll
        for (int i = 0; i < 4; ++i)
#pragma unroll
            for (int j = 0; j < 4; ++j)
                acc[i][j] = __builtin_amdgcn_wmma_f32_16x16x32_bf16(
                    false, a[i].v, false, b[j].v, (short)0, acc[i][j],
                    false, false);
    }

    // C/D layout: VGPR r -> (M = r + 8*hi, N = lo)
#pragma unroll
    for (int i = 0; i < 4; ++i) {
#pragma unroll
        for (int j = 0; j < 4; ++j) {
            const int col = n0 + j * 16 + lo;
#pragma unroll
            for (int r = 0; r < 8; ++r) {
                const int row = m0 + i * 16 + r + hi * 8;
                out[(size_t)row * N + col] = acc[i][j][r];
            }
        }
    }
}

// ---------------------------------------------------------------------------
// RoPE: read raw projection [B*T, nh*128] (f32), write [b, h, t, d] bf16.
// One thread per rotation pair (d, d+64).
// ---------------------------------------------------------------------------
__global__ void rope_to_bf16(const float* __restrict__ src,
                             unsigned short* __restrict__ dst, int nh, int npair) {
    int idx = blockIdx.x * blockDim.x + threadIdx.x;
    if (idx >= npair) return;
    const int i = idx & 63;
    const int h = (idx >> 6) % nh;
    const int t = (idx / (64 * nh)) % Tc;
    const int b = idx / (64 * nh * Tc);
    const float invf = __powf(10000.0f, -(float)i * (1.0f / 64.0f));
    float s, c;
    __sincosf((float)t * invf, &s, &c);
    const size_t so = ((size_t)(b * Tc + t)) * (nh * HDc) + h * HDc + i;
    const float x1 = src[so];
    const float x2 = src[so + 64];
    const size_t dof = (((size_t)(b * nh + h)) * Tc + t) * HDc + i;
    dst[dof]      = f2bf(x1 * c - x2 * s);
    dst[dof + 64] = f2bf(x2 * c + x1 * s);
}

// ---------------------------------------------------------------------------
// V: read raw [B*T, NKV*128] f32, write transposed [b, kvh, d, t] bf16.
// ---------------------------------------------------------------------------
__global__ void v_transpose_bf16(const float* __restrict__ src,
                                 unsigned short* __restrict__ dst, int ntot) {
    int idx = blockIdx.x * blockDim.x + threadIdx.x;
    if (idx >= ntot) return;
    const int t  = idx & (Tc - 1);
    const int d  = (idx / Tc) & (HDc - 1);
    const int kh = (idx / (Tc * HDc)) % NKVc;
    const int b  = idx / (Tc * HDc * NKVc);
    const size_t so = ((size_t)(b * Tc + t)) * (NKVc * HDc) + kh * HDc + d;
    dst[idx] = f2bf(src[so]);
}

// ---------------------------------------------------------------------------
// Flash-style sliding-window attention. One wave per (b, h, 16-query tile).
//   Q:  [b, h,  t, d]  bf16
//   K:  [b, kvh, t, d] bf16
//   Vt: [b, kvh, d, t] bf16
//   out:[b, t, h*128+d] bf16   (A-operand for the Wo GEMM)
// ---------------------------------------------------------------------------
__global__ __launch_bounds__(32) void attn_sliding_window(
        const unsigned short* __restrict__ Q,
        const unsigned short* __restrict__ Km,
        const unsigned short* __restrict__ Vt,
        const unsigned char*  __restrict__ mask,
        unsigned short* __restrict__ Out) {
    __shared__ __align__(16) unsigned short pls[16 * 32];

    const int lane = threadIdx.x;
    const int lo = lane & 15;
    const int hi = lane >> 4;
    const int qb = blockIdx.x;
    const int h  = blockIdx.y;
    const int b  = blockIdx.z;
    const int kvh = h / REPc;
    const int q0 = qb * 16;
    const float scale = 0.08838834764831845f;   // 1/sqrt(128)

    // --- Q A-fragments: 16x128 as 4 chunks of 16x32 -------------------------
    const unsigned short* qrp =
        Q + (((size_t)(b * NHc + h)) * Tc + (q0 + lo)) * HDc;
    Frag16 qf[4];
#pragma unroll
    for (int dc = 0; dc < 4; ++dc) {
        const int d0 = dc * 32 + hi * 8;
        qf[dc].q[0] = *reinterpret_cast<const uint4*>(qrp + d0);
        qf[dc].q[1] = *reinterpret_cast<const uint4*>(qrp + d0 + 16);
    }

    v8f acc[8];
#pragma unroll
    for (int dt = 0; dt < 8; ++dt) acc[dt] = v8f{};
    float mrow[8], lrow[8];
#pragma unroll
    for (int r = 0; r < 8; ++r) { mrow[r] = -3.0e38f; lrow[r] = 0.0f; }

    int kLow = q0 - (WINc - 1); if (kLow < 0) kLow = 0;
    const int kb0   = kLow & ~31;
    const int kbEnd = q0 + 15;

    const unsigned short* kvbase =
        Km + ((size_t)(b * NKVc + kvh)) * Tc * HDc;
    const unsigned short* vtbase =
        Vt + ((size_t)(b * NKVc + kvh)) * HDc * Tc;
    const unsigned char* mrow_base = mask + ((size_t)b * Tc) * Tc;

    for (int kb = kb0; kb <= kbEnd; kb += 32) {
        // ---- S = Q K^T for two 16-key subtiles ----------------------------
        v8f s0 = {}, s1 = {};
        const unsigned short* krow0 = kvbase + (size_t)(kb + lo) * HDc;
        const unsigned short* krow1 = krow0 + 16 * HDc;
#pragma unroll
        for (int dc = 0; dc < 4; ++dc) {
            const int d0 = dc * 32 + hi * 8;
            Frag16 kf;
            kf.q[0] = *reinterpret_cast<const uint4*>(krow0 + d0);
            kf.q[1] = *reinterpret_cast<const uint4*>(krow0 + d0 + 16);
            s0 = __builtin_amdgcn_wmma_f32_16x16x32_bf16(
                false, qf[dc].v, false, kf.v, (short)0, s0, false, false);
            Frag16 kg;
            kg.q[0] = *reinterpret_cast<const uint4*>(krow1 + d0);
            kg.q[1] = *reinterpret_cast<const uint4*>(krow1 + d0 + 16);
            s1 = __builtin_amdgcn_wmma_f32_16x16x32_bf16(
                false, qf[dc].v, false, kg.v, (short)0, s1, false, false);
        }

        // ---- online softmax over this 32-key chunk ------------------------
        float alpha[8];
#pragma unroll
        for (int r = 0; r < 8; ++r) {
            const int qr  = q0 + r + hi * 8;
            const int kc0 = kb + lo;
            const int kc1 = kb + 16 + lo;
            const bool a0 = (kc0 <= qr) && (kc0 >= qr - (WINc - 1)) &&
                            mrow_base[(size_t)qr * Tc + kc0];
            const bool a1 = (kc1 <= qr) && (kc1 >= qr - (WINc - 1)) &&
                            mrow_base[(size_t)qr * Tc + kc1];
            float v0 = a0 ? s0[r] * scale : -3.0e38f;
            float v1 = a1 ? s1[r] * scale : -3.0e38f;
            float red = fmaxf(v0, v1);
#pragma unroll
            for (int off = 1; off < 16; off <<= 1)
                red = fmaxf(red, __shfl_xor(red, off, 32));
            const float nm = fmaxf(mrow[r], red);
            float p0 = a0 ? __expf(v0 - nm) : 0.0f;
            float p1 = a1 ? __expf(v1 - nm) : 0.0f;
            float rs = p0 + p1;
#pragma unroll
            for (int off = 1; off < 16; off <<= 1)
                rs += __shfl_xor(rs, off, 32);
            alpha[r] = __expf(mrow[r] - nm);
            lrow[r]  = lrow[r] * alpha[r] + rs;
            mrow[r]  = nm;
            // stage P (bf16) into LDS in [row][key] order
            const int m_ = r + hi * 8;
            pls[m_ * 32 + lo]      = f2bf(p0);
            pls[m_ * 32 + 16 + lo] = f2bf(p1);
        }
#pragma unroll
        for (int dt = 0; dt < 8; ++dt)
#pragma unroll
            for (int r = 0; r < 8; ++r) acc[dt][r] *= alpha[r];

        asm volatile("s_wait_dscnt 0" ::: "memory");

        // ---- P A-fragment from LDS (A layout: runs at +hi*8 and +16+hi*8) -
        Frag16 pf;
        const int rowb = lo * 32 + hi * 8;
        pf.q[0] = *reinterpret_cast<const uint4*>(&pls[rowb]);
        pf.q[1] = *reinterpret_cast<const uint4*>(&pls[rowb + 16]);

        // ---- acc += P @ V  (8 d-tiles of 16) ------------------------------
#pragma unroll
        for (int dt = 0; dt < 8; ++dt) {
            const unsigned short* vp =
                vtbase + (size_t)(dt * 16 + lo) * Tc + kb + hi * 8;
            Frag16 vf;
            vf.q[0] = *reinterpret_cast<const uint4*>(vp);
            vf.q[1] = *reinterpret_cast<const uint4*>(vp + 16);
            acc[dt] = __builtin_amdgcn_wmma_f32_16x16x32_bf16(
                false, pf.v, false, vf.v, (short)0, acc[dt], false, false);
        }
    }

    // ---- epilogue: normalize and store bf16 [b, t, h*128 + d] --------------
    float inv[8];
#pragma unroll
    for (int r = 0; r < 8; ++r) inv[r] = 1.0f / lrow[r];
#pragma unroll
    for (int dt = 0; dt < 8; ++dt) {
#pragma unroll
        for (int r = 0; r < 8; ++r) {
            const int row = q0 + r + hi * 8;
            Out[((size_t)(b * Tc + row)) * HIDc + h * HDc + dt * 16 + lo] =
                f2bf(acc[dt][r] * inv[r]);
        }
    }
}

// ---------------------------------------------------------------------------
// Host launcher
// ---------------------------------------------------------------------------
extern "C" void kernel_launch(void* const* d_in, const int* in_sizes, int n_in,
                              void* d_out, int out_size, void* d_ws, size_t ws_size,
                              hipStream_t stream) {
    const float*         x    = (const float*)d_in[0];
    const unsigned char* mask = (const unsigned char*)d_in[1];
    const float*         Wq   = (const float*)d_in[2];
    const float*         Wk   = (const float*)d_in[3];
    const float*         Wv   = (const float*)d_in[4];
    const float*         Wo   = (const float*)d_in[5];
    float*               out  = (float*)d_out;

    char* w = (char*)d_ws;
    const size_t MB = 1024ull * 1024ull;
    unsigned short* xbf   = (unsigned short*)(w);              // 16 MB
    float*          qraw  = (float*)(w + 16 * MB);             // 32 MB
    float*          kraw  = (float*)(w + 48 * MB);             //  8 MB
    float*          vraw  = (float*)(w + 56 * MB);             //  8 MB
    unsigned short* qb16  = (unsigned short*)(w + 64 * MB);    // 16 MB
    unsigned short* kb16  = (unsigned short*)(w + 80 * MB);    //  4 MB
    unsigned short* vT16  = (unsigned short*)(w + 84 * MB);    //  4 MB
    unsigned short* wqbf  = (unsigned short*)(w + 88 * MB);    //  8 MB
    unsigned short* wkbf  = (unsigned short*)(w + 96 * MB);    //  2 MB
    unsigned short* wvbf  = (unsigned short*)(w + 98 * MB);    //  2 MB
    unsigned short* wobf  = (unsigned short*)(w + 100 * MB);   //  8 MB
    unsigned short* attnb = (unsigned short*)qraw;             // alias (16 MB)

    const int M = Bc * Tc;                 // 4096 token rows

    // 1) convert activations and all weights to bf16 (weights converted ONCE)
    {
        int n4 = (M * HIDc) / 4;
        cvt_f32_to_bf16<<<(n4 + 255) / 256, 256, 0, stream>>>(x, xbf, n4);
        n4 = (HIDc * HIDc) / 4;
        cvt_f32_to_bf16<<<(n4 + 255) / 256, 256, 0, stream>>>(Wq, wqbf, n4);
        cvt_f32_to_bf16<<<(n4 + 255) / 256, 256, 0, stream>>>(Wo, wobf, n4);
        n4 = (NKVc * HDc * HIDc) / 4;
        cvt_f32_to_bf16<<<(n4 + 255) / 256, 256, 0, stream>>>(Wk, wkbf, n4);
        cvt_f32_to_bf16<<<(n4 + 255) / 256, 256, 0, stream>>>(Wv, wvbf, n4);
    }
    // 2) Q/K/V projections (register-blocked WMMA GEMMs, 64x64 per wave)
    {
        int tiles = (M / 64) * (HIDc / 64);
        wmma_gemm_nt<<<(tiles + 7) / 8, 256, 0, stream>>>(xbf, wqbf, qraw, M, HIDc, HIDc);
        tiles = (M / 64) * ((NKVc * HDc) / 64);
        wmma_gemm_nt<<<(tiles + 7) / 8, 256, 0, stream>>>(xbf, wkbf, kraw, M, NKVc * HDc, HIDc);
        wmma_gemm_nt<<<(tiles + 7) / 8, 256, 0, stream>>>(xbf, wvbf, vraw, M, NKVc * HDc, HIDc);
    }
    // 3) RoPE + layout conversion to bf16
    {
        int npair = Bc * Tc * NHc * 64;
        rope_to_bf16<<<(npair + 255) / 256, 256, 0, stream>>>(qraw, qb16, NHc, npair);
        npair = Bc * Tc * NKVc * 64;
        rope_to_bf16<<<(npair + 255) / 256, 256, 0, stream>>>(kraw, kb16, NKVc, npair);
        const int nv = Bc * NKVc * HDc * Tc;
        v_transpose_bf16<<<(nv + 255) / 256, 256, 0, stream>>>(vraw, vT16, nv);
    }
    // 4) sliding-window flash attention
    {
        dim3 grid(Tc / 16, NHc, Bc);
        attn_sliding_window<<<grid, 32, 0, stream>>>(qb16, kb16, vT16, mask, attnb);
    }
    // 5) output projection -> d_out (f32)
    {
        const int tiles = (M / 64) * (HIDc / 64);
        wmma_gemm_nt<<<(tiles + 7) / 8, 256, 0, stream>>>(attnb, wobf, out, M, HIDc, HIDc);
    }
    (void)in_sizes; (void)n_in; (void)out_size; (void)ws_size;
}
